// DynamicExpertLayer_18451179504161
// MI455X (gfx1250) — matrix operand order repaired and checked by reading, not compiled, not run on hardware
//
#include <hip/hip_runtime.h>
#include <cstdint>

#define N_EXPERTS 8
#define D_MODEL   1024
#define BATCH     4
#define SEQ       2048
#define CAP       512
#define DFF_SUM   30720
#define NT_TOTAL  240   // sum over experts of dff/128
#define LDW       40    // LDS row stride in bf16 (64B data + 16B pad -> conflict-free b128)

typedef __bf16 bf16;
typedef __attribute__((ext_vector_type(16))) bf16  v16bf;
typedef __attribute__((ext_vector_type(8)))  float v8f;
typedef __attribute__((ext_vector_type(4)))  unsigned int u32x4;
typedef __attribute__((ext_vector_type(8)))  int i32x8;
typedef __attribute__((ext_vector_type(4)))  int i32x4;

__constant__ int c_DFF[N_EXPERTS]      = {2048,2560,3072,3584,4096,4608,5120,5632};
__constant__ int c_FF_OFF[N_EXPERTS+1] = {0,2048,4608,7680,11264,15360,19968,25600,30720};
__constant__ int c_NT_CUM[N_EXPERTS+1] = {0,16,36,60,88,120,156,196,240};

#if defined(__HIP_DEVICE_COMPILE__) && __has_builtin(__builtin_amdgcn_tensor_load_to_lds)
#define HAVE_TDM 1
#else
#define HAVE_TDM 0
#endif

// ---------------------------------------------------------------- WMMA helpers
__device__ inline v8f wmma_bf16(v16bf a, v16bf b, v8f c) {
  return __builtin_amdgcn_wmma_f32_16x16x32_bf16(false, a, false, b, (short)0, c,
                                                 false, false);
}

// A fragment (16x32 bf16): lane l -> row = row0+(l&15); 16 contiguous K at
// kb = (l>>4)*16.  Xs rows are LDW bf16 apart in LDS.
__device__ inline v16bf load_a_frag(const bf16* Xs, int row0, int lane) {
  const bf16* p = Xs + (row0 + (lane & 15)) * LDW + ((lane >> 4) * 16);
  v16bf a;
#pragma unroll
  for (int i = 0; i < 8; ++i) { a[i] = p[i]; a[i + 8] = p[i + 8]; }
  return a;
}

// B fragment (32x16 bf16): lane l -> col = col0+(l&15); 16 contiguous K at
// kb2 = (l>>4)*16.  Ws rows (one per output column) are LDW bf16 apart.
__device__ inline v16bf load_b_frag(const bf16* Ws, int col0, int lane) {
  const bf16* p = Ws + (col0 + (lane & 15)) * LDW + ((lane >> 4) * 16);
  v16bf r;
#pragma unroll
  for (int i = 0; i < 16; ++i) r[i] = p[i];
  return r;
}

// ---------------------------------------------------------------- TDM loader
#if HAVE_TDM
// 2D bf16 tile load via Tensor Data Mover, D# packed per CDNA5 ISA section 8.3/8.4.
// LDS padding: pad_enable=1, pad_interval=3 (16 DWORDs = 64B row),
// pad_amount=3 (4 DWORDs = 16B) -> LDS row stride 80B = LDW bf16.
__device__ inline void tdm_load_2d_bf16(uint32_t lds_off, const bf16* gsrc,
                                        uint32_t tensor_d0, uint32_t tensor_d1,
                                        uint32_t tile_d0, uint32_t tile_d1,
                                        uint64_t stride0) {
  uint64_t ga = (uint64_t)(uintptr_t)gsrc;
  u32x4 g0;
  g0[0] = 1u;                                            // count=1
  g0[1] = lds_off;                                       // lds_addr
  g0[2] = (uint32_t)ga;                                  // global_addr lo
  g0[3] = (uint32_t)((ga >> 32) & 0x01FFFFFFu) | (2u << 30);  // addr hi | type=2
  i32x8 g1;
  g1[0] = (int)((1u << 16) |                             // data_size=1 (2 bytes)
                (1u << 20) |                             // pad_enable
                (3u << 22) |                             // pad_interval: 16 DWORDs
                (3u << 25));                             // pad_amount: 4 DWORDs
  g1[1] = (int)((tensor_d0 & 0xFFFFu) << 16);            // tensor_dim0 lo
  g1[2] = (int)((tensor_d0 >> 16) | ((tensor_d1 & 0xFFFFu) << 16));
  g1[3] = (int)((tensor_d1 >> 16) | (tile_d0 << 16));    // tile_dim0
  g1[4] = (int)tile_d1;                                  // tile_dim1 (tile_dim2=0)
  g1[5] = (int)(uint32_t)stride0;                        // tensor_dim0_stride lo
  g1[6] = (int)(uint32_t)((stride0 >> 32) & 0xFFFFu);    // stride0 hi | stride1=0
  g1[7] = 0;
  i32x4 z4; z4[0] = z4[1] = z4[2] = z4[3] = 0;
#if __clang_major__ >= 23
  i32x8 z8;
#pragma unroll
  for (int i = 0; i < 8; ++i) z8[i] = 0;
  __builtin_amdgcn_tensor_load_to_lds(g0, g1, z4, z4, z8, 0);
#else
  __builtin_amdgcn_tensor_load_to_lds(g0, g1, z4, z4, 0);
#endif
}
#endif

__device__ inline void cvt8_store(bf16* d, float4 a, float4 b) {
  d[0] = (bf16)a.x; d[1] = (bf16)a.y; d[2] = (bf16)a.z; d[3] = (bf16)a.w;
  d[4] = (bf16)b.x; d[5] = (bf16)b.y; d[6] = (bf16)b.z; d[7] = (bf16)b.w;
}

// ---------------------------------------------------------------- utility kernels
__global__ __launch_bounds__(256) void zero_f32x4(float4* p, int n4) {
  int i = blockIdx.x * 256 + threadIdx.x;
  if (i < n4) p[i] = make_float4(0.f, 0.f, 0.f, 0.f);
}

// src[R][C] f32  ->  dst[C][R] bf16  (LDS-tiled transpose + convert)
__global__ __launch_bounds__(256) void transpose_f32_bf16(
    const float* __restrict__ src, bf16* __restrict__ dst, int R, int C) {
  __shared__ float tile[32][33];
  const int tx = threadIdx.x & 31;
  const int ty = threadIdx.x >> 5;   // 0..7
  const int c0 = blockIdx.x * 32;
  const int r0 = blockIdx.y * 32;
#pragma unroll
  for (int i = 0; i < 32; i += 8)
    tile[ty + i][tx] = src[(size_t)(r0 + ty + i) * C + (c0 + tx)];
  __syncthreads();
#pragma unroll
  for (int i = 0; i < 32; i += 8)
    dst[(size_t)(c0 + ty + i) * R + (r0 + tx)] = (bf16)tile[tx][ty + i];
}

// ---------------------------------------------------------------- stage 1
// H[b,e][row, n] = silu(X @ W1)[row,n] * (X @ W2)[row,n]   (bf16 out)
// Block tile: 64 (M, gathered tokens) x 128 (N, dff cols), K over D_MODEL,
// double-buffered LDS; weights streamed by the Tensor Data Mover.
__global__ __launch_bounds__(256) void moe_stage1(
    const float* __restrict__ x, const int* __restrict__ tok_idx,
    const bf16* __restrict__ w1t, const bf16* __restrict__ w2t,
    bf16* __restrict__ H) {
  __shared__ bf16 Xs[2][64][LDW];
  __shared__ bf16 W1s[2][128][LDW];
  __shared__ bf16 W2s[2][128][LDW];
  __shared__ int  Ts[64];

  const int tid  = threadIdx.x;
  const int lane = tid & 31;
  const int wv   = tid >> 5;
  const int wm   = wv >> 2;   // 0..1  (M waves)
  const int wn   = wv & 3;    // 0..3  (N waves)

  int idx = blockIdx.x;
  const int b  = idx / (8 * NT_TOTAL);
  int rem      = idx % (8 * NT_TOTAL);
  const int mt = rem / NT_TOTAL;
  int ntg      = rem % NT_TOTAL;
  int e = 0;
  while (ntg >= c_NT_CUM[e + 1]) ++e;
  const int nt  = ntg - c_NT_CUM[e];
  const int dff = c_DFF[e];
  const int ffo = c_FF_OFF[e];

  if (tid < 64)
    Ts[tid] = tok_idx[((b * N_EXPERTS) + e) * CAP + mt * 64 + tid];
  __syncthreads();

  const int gm  = tid >> 2;        // 0..63   row this thread gathers
  const int kko = (tid & 3) * 8;   // 0,8,16,24
  const float* xrow = x + ((size_t)b * SEQ + Ts[gm]) * D_MODEL + kko;

  const bf16* w1p = w1t + (size_t)(ffo + nt * 128) * D_MODEL;  // [dff][1024]
  const bf16* w2p = w2t + (size_t)(ffo + nt * 128) * D_MODEL;

  v8f acc1[2][2] = {};
  v8f acc2[2][2] = {};

  // ---- prologue: fill buffer 0
  {
    const float4* s = (const float4*)xrow;
    cvt8_store(&Xs[0][gm][kko], s[0], s[1]);
  }
#if HAVE_TDM
  if (wv == 0) {
    tdm_load_2d_bf16((uint32_t)(uintptr_t)&W1s[0][0][0], w1p,
                     D_MODEL, (uint32_t)dff, 32, 128, D_MODEL);
    tdm_load_2d_bf16((uint32_t)(uintptr_t)&W2s[0][0][0], w2p,
                     D_MODEL, (uint32_t)dff, 32, 128, D_MODEL);
  }
#else
  {
    const int r  = tid >> 1;
    const int hf = (tid & 1) * 16;
    const uint4* s1 = (const uint4*)(w1p + (size_t)r * D_MODEL + hf);
    const uint4* s2 = (const uint4*)(w2p + (size_t)r * D_MODEL + hf);
    uint4* d1 = (uint4*)&W1s[0][r][hf];
    uint4* d2 = (uint4*)&W2s[0][r][hf];
    d1[0] = s1[0]; d1[1] = s1[1];
    d2[0] = s2[0]; d2[1] = s2[1];
  }
#endif

  const int NCH = D_MODEL / 32;
  for (int i = 0; i < NCH; ++i) {
    const int cur = i & 1;
    const int nxt = cur ^ 1;
#if HAVE_TDM
    if (wv == 0) __builtin_amdgcn_s_wait_tensorcnt(0);
#endif
    __syncthreads();   // publish buffer `cur` (ds stores waited via DScnt here)

    // ---- prefetch chunk i+1 (loads issued now, consumed after the WMMAs)
    const bool pre = (i + 1 < NCH);
    float4 xa, xb;
#if !HAVE_TDM
    uint4 wa0, wa1, wb0, wb1;
    const int fr  = tid >> 1;
    const int fhf = (tid & 1) * 16;
#endif
    if (pre) {
      const int k1 = (i + 1) * 32;
      const float4* s = (const float4*)(xrow + k1);
      xa = s[0]; xb = s[1];
#if HAVE_TDM
      if (wv == 0) {
        tdm_load_2d_bf16((uint32_t)(uintptr_t)&W1s[nxt][0][0], w1p + k1,
                         D_MODEL, (uint32_t)dff, 32, 128, D_MODEL);
        tdm_load_2d_bf16((uint32_t)(uintptr_t)&W2s[nxt][0][0], w2p + k1,
                         D_MODEL, (uint32_t)dff, 32, 128, D_MODEL);
      }
#else
      const uint4* s1 = (const uint4*)(w1p + (size_t)fr * D_MODEL + k1 + fhf);
      const uint4* s2 = (const uint4*)(w2p + (size_t)fr * D_MODEL + k1 + fhf);
      wa0 = s1[0]; wa1 = s1[1]; wb0 = s2[0]; wb1 = s2[1];
#endif
    }

    // ---- compute on buffer `cur`
    v16bf a0 = load_a_frag(&Xs[cur][0][0], wm * 32, lane);
    v16bf a1 = load_a_frag(&Xs[cur][0][0], wm * 32 + 16, lane);
    v16bf p0 = load_b_frag(&W1s[cur][0][0], wn * 32, lane);
    v16bf p1 = load_b_frag(&W1s[cur][0][0], wn * 32 + 16, lane);
    v16bf q0 = load_b_frag(&W2s[cur][0][0], wn * 32, lane);
    v16bf q1 = load_b_frag(&W2s[cur][0][0], wn * 32 + 16, lane);

    acc1[0][0] = wmma_bf16(a0, p0, acc1[0][0]);
    acc1[0][1] = wmma_bf16(a0, p1, acc1[0][1]);
    acc1[1][0] = wmma_bf16(a1, p0, acc1[1][0]);
    acc1[1][1] = wmma_bf16(a1, p1, acc1[1][1]);
    acc2[0][0] = wmma_bf16(a0, q0, acc2[0][0]);
    acc2[0][1] = wmma_bf16(a0, q1, acc2[0][1]);
    acc2[1][0] = wmma_bf16(a1, q0, acc2[1][0]);
    acc2[1][1] = wmma_bf16(a1, q1, acc2[1][1]);

    // ---- drain prefetch into buffer `nxt`
    if (pre) {
      cvt8_store(&Xs[nxt][gm][kko], xa, xb);
#if !HAVE_TDM
      uint4* d1 = (uint4*)&W1s[nxt][fr][fhf];
      uint4* d2 = (uint4*)&W2s[nxt][fr][fhf];
      d1[0] = wa0; d1[1] = wa1;
      d2[0] = wb0; d2[1] = wb1;
#endif
    }
  }

  // ---- epilogue: h = silu(c1) * c2 -> bf16 H
  const size_t Hbase = (size_t)b * CAP * DFF_SUM + (size_t)ffo * CAP;
#pragma unroll
  for (int ti = 0; ti < 2; ++ti)
#pragma unroll
    for (int tj = 0; tj < 2; ++tj) {
      const int col = nt * 128 + wn * 32 + tj * 16 + (lane & 15);
      const int r0  = mt * 64 + wm * 32 + ti * 16 + ((lane >> 4) << 3);
#pragma unroll
      for (int i = 0; i < 8; ++i) {
        float g = acc1[ti][tj][i];
        float u = acc2[ti][tj][i];
        float h = (g / (1.0f + __expf(-g))) * u;
        H[Hbase + (size_t)(r0 + i) * dff + col] = (bf16)h;
      }
    }
}

// ---------------------------------------------------------------- stage 2
// out = H @ W3;  write expert_outs and atomically scatter out*w into output.
__global__ __launch_bounds__(256) void moe_stage2(
    const bf16* __restrict__ H, const bf16* __restrict__ w3t,
    const int* __restrict__ tok_idx, const float* __restrict__ ew,
    float* __restrict__ out_main, float* __restrict__ expert_outs) {
  __shared__ bf16  As[2][64][LDW];
  __shared__ bf16  W3s[2][128][LDW];
  __shared__ int   Ts[64];
  __shared__ float Wt[64];

  const int tid  = threadIdx.x;
  const int lane = tid & 31;
  const int wv   = tid >> 5;
  const int wm   = wv >> 2;
  const int wn   = wv & 3;

  int idx = blockIdx.x;
  const int b  = idx >> 9;        // 8 experts * 8 mtiles * 8 ntiles = 512
  int rem      = idx & 511;
  const int e  = rem >> 6;
  rem &= 63;
  const int mt = rem >> 3;
  const int nt = rem & 7;
  const int dff = c_DFF[e];
  const int ffo = c_FF_OFF[e];

  if (tid < 64) {
    int o = ((b * N_EXPERTS) + e) * CAP + mt * 64 + tid;
    Ts[tid] = tok_idx[o];
    Wt[tid] = ew[o];
  }
  __syncthreads();

  const int gm  = tid >> 2;
  const int kko = (tid & 3) * 8;
  const size_t Hbase = (size_t)b * CAP * DFF_SUM + (size_t)ffo * CAP;
  const bf16* hrow = H + Hbase + (size_t)(mt * 64 + gm) * dff + kko;
  const bf16* w3p  = w3t + (size_t)ffo * D_MODEL + (size_t)(nt * 128) * dff;  // [1024][dff]

  v8f acc[2][2] = {};

  // ---- prologue: fill buffer 0
  ((uint4*)&As[0][gm][kko])[0] = ((const uint4*)hrow)[0];
#if HAVE_TDM
  if (wv == 0)
    tdm_load_2d_bf16((uint32_t)(uintptr_t)&W3s[0][0][0], w3p,
                     (uint32_t)dff, D_MODEL, 32, 128, (uint64_t)dff);
#else
  {
    const int r  = tid >> 1;
    const int hf = (tid & 1) * 16;
    const uint4* s = (const uint4*)(w3p + (size_t)r * dff + hf);
    uint4* d = (uint4*)&W3s[0][r][hf];
    d[0] = s[0]; d[1] = s[1];
  }
#endif

  const int NCH = dff / 32;
  for (int i = 0; i < NCH; ++i) {
    const int cur = i & 1;
    const int nxt = cur ^ 1;
#if HAVE_TDM
    if (wv == 0) __builtin_amdgcn_s_wait_tensorcnt(0);
#endif
    __syncthreads();

    const bool pre = (i + 1 < NCH);
    uint4 h4;
#if !HAVE_TDM
    uint4 w40, w41;
    const int fr  = tid >> 1;
    const int fhf = (tid & 1) * 16;
#endif
    if (pre) {
      const int k1 = (i + 1) * 32;
      h4 = ((const uint4*)(hrow + k1))[0];
#if HAVE_TDM
      if (wv == 0)
        tdm_load_2d_bf16((uint32_t)(uintptr_t)&W3s[nxt][0][0], w3p + k1,
                         (uint32_t)dff, D_MODEL, 32, 128, (uint64_t)dff);
#else
      const uint4* s = (const uint4*)(w3p + (size_t)fr * dff + k1 + fhf);
      w40 = s[0]; w41 = s[1];
#endif
    }

    v16bf a0 = load_a_frag(&As[cur][0][0], wm * 32, lane);
    v16bf a1 = load_a_frag(&As[cur][0][0], wm * 32 + 16, lane);
    v16bf b0 = load_b_frag(&W3s[cur][0][0], wn * 32, lane);
    v16bf b1 = load_b_frag(&W3s[cur][0][0], wn * 32 + 16, lane);

    acc[0][0] = wmma_bf16(a0, b0, acc[0][0]);
    acc[0][1] = wmma_bf16(a0, b1, acc[0][1]);
    acc[1][0] = wmma_bf16(a1, b0, acc[1][0]);
    acc[1][1] = wmma_bf16(a1, b1, acc[1][1]);

    if (pre) {
      ((uint4*)&As[nxt][gm][kko])[0] = h4;
#if !HAVE_TDM
      uint4* d = (uint4*)&W3s[nxt][fr][fhf];
      d[0] = w40; d[1] = w41;
#endif
    }
  }

  float* eo = expert_outs + (size_t)e * (BATCH * CAP * D_MODEL);
#pragma unroll
  for (int ti = 0; ti < 2; ++ti)
#pragma unroll
    for (int tj = 0; tj < 2; ++tj) {
      const int col = nt * 128 + wn * 32 + tj * 16 + (lane & 15);
      const int rl0 = wm * 32 + ti * 16 + ((lane >> 4) << 3);
#pragma unroll
      for (int i = 0; i < 8; ++i) {
        const int rl  = rl0 + i;
        const int row = mt * 64 + rl;
        float v = acc[ti][tj][i];
        eo[((size_t)b * CAP + row) * D_MODEL + col] = v;
        atomicAdd(&out_main[((size_t)b * SEQ + Ts[rl]) * D_MODEL + col],
                  v * Wt[rl]);
      }
    }
}

// ---------------------------------------------------------------- launch
extern "C" void kernel_launch(void* const* d_in, const int* in_sizes, int n_in,
                              void* d_out, int out_size, void* d_ws, size_t ws_size,
                              hipStream_t stream) {
  static const int hDFF[N_EXPERTS]    = {2048,2560,3072,3584,4096,4608,5120,5632};
  static const int hFFO[N_EXPERTS+1]  = {0,2048,4608,7680,11264,15360,19968,25600,30720};

  const float* x  = (const float*)d_in[0];
  const float* ew = (const float*)d_in[1];
  const int*   ti = (const int*)d_in[2];

  bf16* w1t = (bf16*)d_ws;                               // [sum dff][1024] bf16
  bf16* w2t = w1t + (size_t)DFF_SUM * D_MODEL;           // [sum dff][1024] bf16
  bf16* w3t = w2t + (size_t)DFF_SUM * D_MODEL;           // per-e [1024][dff] bf16
  bf16* H   = w3t + (size_t)DFF_SUM * D_MODEL;           // per (b,e) [512][dff] bf16

  float* out_main = (float*)d_out;
  float* eouts    = out_main + (size_t)BATCH * SEQ * D_MODEL;

  // Phase 0: weight convert/transpose + zero-init of scatter-add target.
  for (int e = 0; e < N_EXPERTS; ++e) {
    const int dff = hDFF[e];
    dim3 g12(dff / 32, D_MODEL / 32);
    dim3 g3(D_MODEL / 32, dff / 32);
    transpose_f32_bf16<<<g12, 256, 0, stream>>>(
        (const float*)d_in[3 + e],  w1t + (size_t)hFFO[e] * D_MODEL, D_MODEL, dff);
    transpose_f32_bf16<<<g12, 256, 0, stream>>>(
        (const float*)d_in[11 + e], w2t + (size_t)hFFO[e] * D_MODEL, D_MODEL, dff);
    transpose_f32_bf16<<<g3, 256, 0, stream>>>(
        (const float*)d_in[19 + e], w3t + (size_t)hFFO[e] * D_MODEL, dff, D_MODEL);
  }
  const int n4 = (BATCH * SEQ * D_MODEL) / 4;
  zero_f32x4<<<(n4 + 255) / 256, 256, 0, stream>>>((float4*)out_main, n4);

  // Phase 1: H = silu(X@W1) * (X@W2)
  moe_stage1<<<BATCH * 8 * NT_TOTAL, 256, 0, stream>>>(x, ti, w1t, w2t, H);

  // Phase 2: out = H@W3, expert_outs + weighted scatter-add.
  moe_stage2<<<BATCH * N_EXPERTS * 8 * 8, 256, 0, stream>>>(H, w3t, ti, ew,
                                                            out_main, eouts);
}
